// TemporalAttention_31482110279897
// MI455X (gfx1250) — matrix operand order repaired and checked by reading, not compiled, not run on hardware
//
#include <hip/hip_runtime.h>

typedef __bf16 bf16_t;
typedef __attribute__((ext_vector_type(8)))  bf16_t bf16x8;
typedef __attribute__((ext_vector_type(16))) bf16_t bf16x16;
typedef __attribute__((ext_vector_type(4)))  float  f32x4;
typedef __attribute__((ext_vector_type(8)))  float  v8f;
typedef __attribute__((ext_vector_type(2)))  unsigned int u32x2;
typedef __attribute__((ext_vector_type(4)))  unsigned int u32x4;
typedef __attribute__((ext_vector_type(4)))  int          i32x4;
typedef __attribute__((ext_vector_type(8)))  int          i32x8;

constexpr int kHeads = 8;
constexpr int kDh    = 64;
constexpr int kF     = 16;
constexpr int kC     = 512;

// cheap fp32 -> bf16 (round-half-up): one v_add, stores fold into d16_hi
__device__ __forceinline__ bf16_t f2bf(float f) {
  unsigned u = __builtin_bit_cast(unsigned, f) + 0x8000u;
  unsigned short hbits = (unsigned short)(u >> 16);
  return __builtin_bit_cast(bf16_t, hbits);
}

// pack two fp32 into two bf16 in one dword via v_perm_b32
__device__ __forceinline__ unsigned pack_bf2(float lo, float hi) {
  unsigned ulo = __builtin_bit_cast(unsigned, lo) + 0x8000u;
  unsigned uhi = __builtin_bit_cast(unsigned, hi) + 0x8000u;
  // result bytes = [ulo.b2, ulo.b3, uhi.b2, uhi.b3]
  return __builtin_amdgcn_perm(uhi, ulo, 0x07060302u);
}

__device__ __forceinline__ bf16x16 cat8(bf16x8 lo, bf16x8 hi) {
  return __builtin_shufflevector(lo, hi, 0,1,2,3,4,5,6,7,8,9,10,11,12,13,14,15);
}

__device__ __forceinline__ v8f wmma_bf16(bf16x16 a, bf16x16 b, v8f c) {
  return __builtin_amdgcn_wmma_f32_16x16x32_bf16(false, a, false, b, (short)0, c,
                                                 false, false);
}

// ---------------------------------------------------------------------------
// TDM: async 2D tile DMA (global bf16 -> LDS), D# per CDNA5 ISA.
// Issue from ONE wave only (EXEC is ignored by TENSOR_LOAD_TO_LDS).
// This toolchain exposes the 6-arg builtin: (g0, g1, g2, g3, g_extra, cpol).
// ---------------------------------------------------------------------------
__device__ __forceinline__ void tdm_load_2d_bf16(
    unsigned lds_off, const void* gptr, int tile_elems_x, int tile_rows,
    int tensor_dim0, int row_stride_elems) {
  unsigned long long ga = (unsigned long long)(size_t)gptr;
  u32x4 g0;
  g0[0] = 1u;                                           // count=1 (valid D#)
  g0[1] = lds_off;                                      // lds_addr (bytes)
  g0[2] = (unsigned)(ga & 0xffffffffu);                 // global_addr[31:0]
  g0[3] = (unsigned)((ga >> 32) & 0x1ffffffu)           // global_addr[56:32]
          | (2u << 30);                                 // type=2 ("image")
  i32x8 g1;
  g1[0] = 1 << 16;                                      // data_size=1 (2B)
  g1[1] = (tensor_dim0 & 0xffff) << 16;                 // tensor_dim0[15:0]
  g1[2] = (int)(((unsigned)tensor_dim0 >> 16)           // tensor_dim0[31:16]
          | ((unsigned)(tile_rows & 0xffff) << 16));    // tensor_dim1[15:0]
  g1[3] = (tile_elems_x & 0xffff) << 16;                // tile_dim0
  g1[4] = tile_rows & 0xffff;                           // tile_dim1
  g1[5] = row_stride_elems;                             // tensor_dim0_stride
  g1[6] = 0;
  g1[7] = 0;
  i32x4 z4 = {};
  i32x8 z8 = {};
  __builtin_amdgcn_tensor_load_to_lds(g0, g1, z4, z4, z8, 0);
}

// ---------------------------------------------------------------------------
// Kernel 1: fused Q/K/V projection. GEMM M=tokens, N=512, K=512, z = which.
// Output: Q,K as [b][h][f][d] bf16 ; V transposed as [b][h][d][f] bf16.
// ---------------------------------------------------------------------------
__global__ __launch_bounds__(256) void qkv_kernel(
    const float* __restrict__ X, const float* __restrict__ Wq,
    const float* __restrict__ Wk, const float* __restrict__ Wv,
    bf16_t* __restrict__ q_ws, bf16_t* __restrict__ k_ws,
    bf16_t* __restrict__ v_ws) {
  __shared__ __align__(16) bf16_t As[64][32];    // A tile, row-major [m][k]
  __shared__ __align__(16) bf16_t Bt[128][32];   // B tile transposed [n][k]
  __shared__ __align__(16) bf16_t Ct[64][128];   // staged C tile (bf16)

  const int t    = threadIdx.x;
  const int lane = t & 31;
  const int wid  = t >> 5;
  const int hf   = lane >> 4;
  const int l16  = lane & 15;

  const int M0 = blockIdx.x * 64;
  const int N0 = blockIdx.y * 128;
  const int which = blockIdx.z;
  const float* W = (which == 0) ? Wq : (which == 1) ? Wk : Wv;

  const int wm = (wid >> 1) * 16;
  const int wn = (wid & 1) * 64;

  v8f acc[4] = {};

  for (int k0 = 0; k0 < kC; k0 += 32) {
    // stage A: X[M0..+63][k0..+31] fp32 -> bf16 (streaming: non-temporal)
    #pragma unroll
    for (int i = 0; i < 2; i++) {
      int idx  = t + i * 256;
      int row  = idx >> 3;
      int col4 = idx & 7;
      f32x4 xv = __builtin_nontemporal_load(
          (const f32x4*)&X[(size_t)(M0 + row) * kC + (k0 + col4 * 4)]);
      u32x2 pk;
      pk[0] = pack_bf2(xv[0], xv[1]);
      pk[1] = pack_bf2(xv[2], xv[3]);
      *(u32x2*)&As[row][col4 * 4] = pk;
    }
    // stage B transposed, two k-rows per thread packed into one dword
    #pragma unroll
    for (int i = 0; i < 8; i++) {
      int p   = t + i * 256;          // 2048 pairs over 32x128
      int nc  = p & 127;
      int kr  = (p >> 7) * 2;
      float w0 = W[(size_t)(k0 + kr) * kC + (N0 + nc)];
      float w1 = W[(size_t)(k0 + kr + 1) * kC + (N0 + nc)];
      *(unsigned*)&Bt[nc][kr] = pack_bf2(w0, w1);
    }
    __syncthreads();

    bf16x8 alo = *(const bf16x8*)&As[wm + l16][hf * 8];
    bf16x8 ahi = *(const bf16x8*)&As[wm + l16][16 + hf * 8];
    bf16x16 afrag = cat8(alo, ahi);
    #pragma unroll
    for (int nn = 0; nn < 4; nn++) {
      int n = wn + nn * 16 + l16;
      bf16x8 blo = *(const bf16x8*)&Bt[n][hf * 8];
      bf16x8 bhi = *(const bf16x8*)&Bt[n][16 + hf * 8];
      acc[nn] = wmma_bf16(afrag, cat8(blo, bhi), acc[nn]);
    }
    __syncthreads();
  }

  // C fragment -> LDS (element (m = r + 8*hf, n = l16))
  #pragma unroll
  for (int nn = 0; nn < 4; nn++)
    #pragma unroll
    for (int r = 0; r < 8; r++)
      Ct[wm + r + 8 * hf][wn + nn * 16 + l16] = f2bf(acc[nn][r]);
  __syncthreads();

  if (which < 2) {
    bf16_t* outp = (which == 0) ? q_ws : k_ws;
    #pragma unroll
    for (int i = 0; i < 4; i++) {
      int cc  = t + i * 256;
      int row = cc >> 4;
      int seg = cc & 15;
      int token = M0 + row;
      int bb = token >> 4, ff = token & 15;
      int cg = N0 + seg * 8;
      int hh = cg >> 6, dd = cg & 63;
      bf16x8 vdat = *(const bf16x8*)&Ct[row][seg * 8];
      *(bf16x8*)&outp[(((size_t)(bb * kHeads + hh) * kF) + ff) * kDh + dd] = vdat;
    }
  } else {
    // V transposed: [b][h][d][f]
    #pragma unroll
    for (int i = 0; i < 2; i++) {
      int u  = t + i * 256;
      int bg = u >> 7;
      int c  = u & 127;
      int bb = (M0 + bg * 16) >> 4;
      int hh = (N0 + c) >> 6, dd = (N0 + c) & 63;
      bf16x8 p0, p1;
      #pragma unroll
      for (int fr = 0; fr < 8; fr++) p0[fr] = Ct[bg * 16 + fr][c];
      #pragma unroll
      for (int fr = 0; fr < 8; fr++) p1[fr] = Ct[bg * 16 + 8 + fr][c];
      bf16_t* dst = &v_ws[((size_t)(bb * kHeads + hh) * kDh + dd) * kF];
      *(bf16x8*)&dst[0] = p0;
      *(bf16x8*)&dst[8] = p1;
    }
  }
}

// ---------------------------------------------------------------------------
// T5 relative position bias (NUM_BUCKETS=32, MAX_DISTANCE=128)
// ---------------------------------------------------------------------------
__device__ __forceinline__ float rel_bias(int i, int j, int h,
                                          const float* __restrict__ table) {
  int n = i - j;
  int ret = 0;
  if (n < 0) { ret = 16; n = -n; }
  int bucket;
  if (n < 8) {
    bucket = ret + n;
  } else {
    float lg = __logf((float)n * 0.125f) * (8.0f / 2.7725887f);
    int vl = 8 + (int)lg;
    bucket = ret + (vl < 15 ? vl : 15);
  }
  return table[bucket * kHeads + h];
}

// ---------------------------------------------------------------------------
// Kernel 2: attention. One wave per (b,h).
// ---------------------------------------------------------------------------
__global__ __launch_bounds__(256) void attn_kernel(
    const bf16_t* __restrict__ q_ws, const bf16_t* __restrict__ k_ws,
    const bf16_t* __restrict__ v_ws, const float* __restrict__ table,
    bf16_t* __restrict__ attn_ws) {
  __shared__ __align__(16) bf16_t Ps[8][16][16];
  __shared__ __align__(16) bf16_t Os[8][16][64];

  const int t    = threadIdx.x;
  const int lane = t & 31;
  const int wid  = t >> 5;
  const int hf   = lane >> 4;
  const int l16  = lane & 15;

  const int bh = blockIdx.x * 8 + wid;
  const int b  = bh >> 3;
  const int h  = bh & 7;

  const bf16_t* Q  = q_ws + (size_t)bh * kF * kDh;   // [f][d]
  const bf16_t* K  = k_ws + (size_t)bh * kF * kDh;   // [f][d]
  const bf16_t* Vt = v_ws + (size_t)bh * kDh * kF;   // [d][f]

  v8f s = {};
  #pragma unroll
  for (int d0 = 0; d0 < kDh; d0 += 32) {
    bf16x8 alo = *(const bf16x8*)&Q[l16 * kDh + d0 + hf * 8];
    bf16x8 ahi = *(const bf16x8*)&Q[l16 * kDh + d0 + 16 + hf * 8];
    bf16x8 blo = *(const bf16x8*)&K[l16 * kDh + d0 + hf * 8];
    bf16x8 bhi = *(const bf16x8*)&K[l16 * kDh + d0 + 16 + hf * 8];
    s = wmma_bf16(cat8(alo, ahi), cat8(blo, bhi), s);
  }

  float p[8];
  #pragma unroll
  for (int r = 0; r < 8; r++) {
    int i = r + 8 * hf;
    p[r] = s[r] * 0.125f + rel_bias(i, l16, h, table);
  }
  #pragma unroll
  for (int r = 0; r < 8; r++) {
    float mx = p[r];
    #pragma unroll
    for (int msk = 1; msk < 16; msk <<= 1)
      mx = fmaxf(mx, __shfl_xor(mx, msk, 32));
    float e = __expf(p[r] - mx);
    float sm = e;
    #pragma unroll
    for (int msk = 1; msk < 16; msk <<= 1)
      sm += __shfl_xor(sm, msk, 32);
    p[r] = e / sm;
  }

  #pragma unroll
  for (int r = 0; r < 8; r++) Ps[wid][r + 8 * hf][l16] = f2bf(p[r]);
  __syncthreads();

  bf16x8 plo = *(const bf16x8*)&Ps[wid][l16][hf * 8];
  bf16x8 z8  = {};
  bf16x16 pf = cat8(plo, z8);

  #pragma unroll
  for (int d0 = 0; d0 < kDh; d0 += 16) {
    bf16x8 v0 = *(const bf16x8*)&Vt[(d0 + l16) * kF + hf * 8];
    bf16x16 vf = cat8(v0, z8);
    v8f o = {};
    o = wmma_bf16(pf, vf, o);
    #pragma unroll
    for (int r = 0; r < 8; r++) Os[wid][r + 8 * hf][d0 + l16] = f2bf(o[r]);
  }
  __syncthreads();

  bf16_t* dst = &attn_ws[((size_t)(b * kF + l16)) * (kHeads * kDh) +
                         h * kDh + hf * 32];
  const bf16_t* src = &Os[wid][l16][hf * 32];
  #pragma unroll
  for (int i = 0; i < 4; i++)
    *(bf16x8*)&dst[i * 8] = *(const bf16x8*)&src[i * 8];
}

// ---------------------------------------------------------------------------
// Kernel 3: output projection  out = attn @ Wo + bo  (fp32 out).
// A tile arrives via Tensor Data Mover DMA (global bf16 -> LDS).
// ---------------------------------------------------------------------------
__global__ __launch_bounds__(256) void proj_kernel(
    const bf16_t* __restrict__ A, const float* __restrict__ Wo,
    const float* __restrict__ bo, float* __restrict__ outp) {
  __shared__ __align__(16) bf16_t As[64][32];
  __shared__ __align__(16) bf16_t Bt[128][32];

  const int t    = threadIdx.x;
  const int lane = t & 31;
  const int wid  = t >> 5;
  const int hf   = lane >> 4;
  const int l16  = lane & 15;

  const int M0 = blockIdx.x * 64;
  const int N0 = blockIdx.y * 128;
  const int wm = (wid >> 1) * 16;
  const int wn = (wid & 1) * 64;

  const unsigned as_lds = (unsigned)(size_t)(&As[0][0]);

  v8f acc[4] = {};

  for (int k0 = 0; k0 < kC; k0 += 32) {
    // stage B transposed (paired pack, one b32 DS store per 2 elements)
    #pragma unroll
    for (int i = 0; i < 8; i++) {
      int p  = t + i * 256;
      int nc = p & 127;
      int kr = (p >> 7) * 2;
      float w0 = Wo[(size_t)(k0 + kr) * kC + (N0 + nc)];
      float w1 = Wo[(size_t)(k0 + kr + 1) * kC + (N0 + nc)];
      *(unsigned*)&Bt[nc][kr] = pack_bf2(w0, w1);
    }
    // stage A via TDM: 64 rows x 32 bf16, row stride kC elems.
    // wave-uniform branch: TDM ignores EXEC, so only wave 0 may issue it.
    if (wid == 0) {
      tdm_load_2d_bf16(as_lds, &A[(size_t)M0 * kC + k0],
                       /*tile_x=*/32, /*tile_rows=*/64,
                       /*tensor_dim0=*/kC, /*row_stride=*/kC);
      __builtin_amdgcn_s_wait_tensorcnt(0);
    }
    __syncthreads();

    bf16x8 alo = *(const bf16x8*)&As[wm + l16][hf * 8];
    bf16x8 ahi = *(const bf16x8*)&As[wm + l16][16 + hf * 8];
    bf16x16 afrag = cat8(alo, ahi);
    #pragma unroll
    for (int nn = 0; nn < 4; nn++) {
      int n = wn + nn * 16 + l16;
      bf16x8 blo = *(const bf16x8*)&Bt[n][hf * 8];
      bf16x8 bhi = *(const bf16x8*)&Bt[n][16 + hf * 8];
      acc[nn] = wmma_bf16(afrag, cat8(blo, bhi), acc[nn]);
    }
    __syncthreads();
  }

  // fp32 store straight from C fragment, non-temporal (streaming output)
  #pragma unroll
  for (int nn = 0; nn < 4; nn++) {
    int n = N0 + wn + nn * 16 + l16;
    float bias = bo[n];
    #pragma unroll
    for (int r = 0; r < 8; r++) {
      int m = M0 + wm + r + 8 * hf;
      __builtin_nontemporal_store(acc[nn][r] + bias,
                                  &outp[(size_t)m * kC + n]);
    }
  }
}

// ---------------------------------------------------------------------------
extern "C" void kernel_launch(void* const* d_in, const int* in_sizes, int n_in,
                              void* d_out, int out_size, void* d_ws,
                              size_t ws_size, hipStream_t stream) {
  const float* X     = (const float*)d_in[0];
  const float* Wq    = (const float*)d_in[1];
  const float* Wk    = (const float*)d_in[2];
  const float* Wv    = (const float*)d_in[3];
  const float* Wo    = (const float*)d_in[4];
  const float* bo    = (const float*)d_in[5];
  const float* table = (const float*)d_in[6];
  float* outp = (float*)d_out;

  const int ntok = in_sizes[0] / kC;   // B * F tokens
  const int nb   = ntok / kF;          // B

  bf16_t* q_ws = (bf16_t*)d_ws;
  bf16_t* k_ws = q_ws + (size_t)ntok * kC;
  bf16_t* v_ws = k_ws + (size_t)ntok * kC;
  bf16_t* a_ws = v_ws + (size_t)ntok * kC;

  qkv_kernel<<<dim3(ntok / 64, kC / 128, 3), 256, 0, stream>>>(
      X, Wq, Wk, Wv, q_ws, k_ws, v_ws);
  attn_kernel<<<dim3(nb), 256, 0, stream>>>(q_ws, k_ws, v_ws, table, a_ws);
  proj_kernel<<<dim3(ntok / 64, kC / 128), 256, 0, stream>>>(a_ws, Wo, bo, outp);
}
